// AttentionHead_54305566491101
// MI455X (gfx1250) — matrix operand order repaired and checked by reading, not compiled, not run on hardware
//
#include <hip/hip_runtime.h>
#include <hip/hip_bf16.h>

// ---------------------------------------------------------------------------
// AttentionHead with 5x5 toroidal locality-masked projections, gfx1250 WMMA.
// B=8, S=2048, H=1024, side=32, window 5x5 (25 taps per output unit).
// ---------------------------------------------------------------------------

#define B_  8
#define S_  2048
#define H_  1024
#define SIDE 32
#define HALFW 2               // width 5 -> half 2
#define INV_SQRT_H 0.03125f   // 1/sqrt(1024)

typedef __attribute__((ext_vector_type(16))) _Float16 v16h;
typedef __attribute__((ext_vector_type(8)))  _Float16 v8h;
typedef __attribute__((ext_vector_type(8)))  float    v8f;

// ---------------------------------------------------------------------------
// Fragment loaders following CDNA5 ISA 7.12.2 VGPR layouts (wave32).
// A (16x32 f16, row-major src [m][k]):
//   lane L: M = L%16 ; kb = (L>=16)?8:0 ; elems 0..7 -> K=kb+e ; 8..15 -> K=kb+16+e
// B (32x16 f16, we need B[k][n] = src[n][k], src row-major [n][k]):
//   lane L: N = L%16 ; kb = (L>=16)?8:0 ; elems 0..7 -> K=kb+e ; 8..15 -> K=kb+16+e
// C/D (16x16 f32): lane L: N = L%16 ; vgpr i -> M = i + 8*(L>=16)
// ---------------------------------------------------------------------------
__device__ __forceinline__ v16h load_a_f16(const _Float16* __restrict__ base,
                                           long row0, int ld, int k0, int lane) {
  const int m  = lane & 15;
  const int kb = (lane & 16) ? 8 : 0;
  const _Float16* p = base + (long)(row0 + m) * ld + k0 + kb;
  v8h lo = *(const v8h*)(p);
  v8h hi = *(const v8h*)(p + 16);
  v16h a;
#pragma unroll
  for (int e = 0; e < 8; ++e) { a[e] = lo[e]; a[8 + e] = hi[e]; }
  return a;
}

__device__ __forceinline__ v16h load_a_f32cvt(const float* __restrict__ base,
                                              long row0, int ld, int k0, int lane) {
  const int m  = lane & 15;
  const int kb = (lane & 16) ? 8 : 0;
  const float* p = base + (long)(row0 + m) * ld + k0 + kb;
  v16h a;
#pragma unroll
  for (int e = 0; e < 8; ++e) {
    a[e]     = (_Float16)p[e];
    a[8 + e] = (_Float16)p[16 + e];
  }
  return a;
}

__device__ __forceinline__ v16h load_b_f16(const _Float16* __restrict__ base,
                                           long nrow0, int ld, int k0, int lane) {
  const int n  = lane & 15;
  const int kb = (lane & 16) ? 8 : 0;
  const _Float16* p = base + (long)(nrow0 + n) * ld + k0 + kb;
  v8h b0 = *(const v8h*)(p);
  v8h b1 = *(const v8h*)(p + 16);
  v16h b;
#pragma unroll
  for (int e = 0; e < 8; ++e) { b[e] = b0[e]; b[8 + e] = b1[e]; }
  return b;
}

// ---------------------------------------------------------------------------
// K1: masked projections as 25-tap toroidal stencils (sparse => VALU).
// One block per (b,s) row. Emits f16 q,k row-major [B*S][H] and v transposed
// [B][H][S] so the K3 B-fragment is two contiguous 8-half loads per lane.
// ---------------------------------------------------------------------------
__global__ __launch_bounds__(256)
void proj_stencil_kernel(const float* __restrict__ x,
                         const float* __restrict__ Wq, const float* __restrict__ bq,
                         const float* __restrict__ Wk, const float* __restrict__ bk,
                         const float* __restrict__ Wv, const float* __restrict__ bv,
                         const float* __restrict__ Wl,
                         _Float16* __restrict__ q16,
                         _Float16* __restrict__ k16,
                         _Float16* __restrict__ vT16) {
  __shared__ float xs[H_];
  __shared__ float q0s[H_];

  const int tid = threadIdx.x;
  const long row = blockIdx.x;           // 0 .. B*S-1
  const int  b   = (int)(row / S_);
  const int  s   = (int)(row % S_);

  // stage x row into LDS
  for (int i = tid; i < H_; i += 256) xs[i] = x[row * H_ + i];
  __syncthreads();

  // q0 / k / v stencils (4 channels per thread)
#pragma unroll
  for (int p = 0; p < 4; ++p) {
    const int o  = tid + p * 256;
    const int r  = o >> 5;
    const int c  = o & 31;
    const long wo = (long)o * H_;
    float aq = bq[o], ak = bk[o], av = bv[o];
#pragma unroll
    for (int dr = -HALFW; dr <= HALFW; ++dr) {
      const int rr = (r + dr) & (SIDE - 1);
#pragma unroll
      for (int dc = -HALFW; dc <= HALFW; ++dc) {
        const int j  = (rr << 5) | ((c + dc) & (SIDE - 1));
        const float xv = xs[j];
        aq = fmaf(Wq[wo + j], xv, aq);
        ak = fmaf(Wk[wo + j], xv, ak);
        av = fmaf(Wv[wo + j], xv, av);
      }
    }
    q0s[o] = aq;
    k16[row * H_ + o] = (_Float16)ak;
    vT16[((long)b * H_ + o) * S_ + s] = (_Float16)av;
  }
  __syncthreads();

  // second stencil: q = q0 @ (Wl*mask)^T
#pragma unroll
  for (int p = 0; p < 4; ++p) {
    const int o  = tid + p * 256;
    const int r  = o >> 5;
    const int c  = o & 31;
    const long wo = (long)o * H_;
    float aq = 0.f;
#pragma unroll
    for (int dr = -HALFW; dr <= HALFW; ++dr) {
      const int rr = (r + dr) & (SIDE - 1);
#pragma unroll
      for (int dc = -HALFW; dc <= HALFW; ++dc) {
        const int j = (rr << 5) | ((c + dc) & (SIDE - 1));
        aq = fmaf(Wl[wo + j], q0s[j], aq);
      }
    }
    q16[row * H_ + o] = (_Float16)aq;
  }
}

// ---------------------------------------------------------------------------
// K2: scores = q k^T / 32, rowwise softmax, write probs (f32).
// Block = 32 rows x 1024 cols; 8 waves, each owning a 128-col strip.
// Per wave: 2 A fragments (rows m..m+15, m+16..m+31) x 8 column tiles
// => 16 f32 accumulators; every B fragment feeds TWO wmma (A reuse), and
// B fragments are software-pipelined (b_nxt issued before wmma on b_cur).
// ---------------------------------------------------------------------------
__global__ __launch_bounds__(256)
void scores_softmax_kernel(const _Float16* __restrict__ q16,
                           const _Float16* __restrict__ k16,
                           float* __restrict__ probs) {
  __shared__ float red[32 * 256];   // 32 KB
  __shared__ float rmax[32];
  __shared__ float rsum[32];

  const int tid  = threadIdx.x;
  const int lane = tid & 31;
  const int w    = tid >> 5;          // wave 0..7
  const int hi   = (lane >> 4) & 1;   // lane-half
  const int b    = blockIdx.y;
  const int m0   = blockIdx.x * 32;   // row tile within batch
  const long qrow0 = (long)b * S_ + m0;
  const long krow0 = (long)b * S_ + (long)w * 128;

  v8f acc[16];                        // [t]: rows m0..+15 ; [8+t]: rows m0+16..+31
  const v8f vz = {};
#pragma unroll
  for (int t = 0; t < 16; ++t) acc[t] = vz;

  for (int k0 = 0; k0 < H_; k0 += 32) {
    const v16h a0 = load_a_f16(q16, qrow0,      H_, k0, lane);
    const v16h a1 = load_a_f16(q16, qrow0 + 16, H_, k0, lane);
    v16h bcur = load_b_f16(k16, krow0, H_, k0, lane);
#pragma unroll
    for (int t = 0; t < 8; ++t) {
      v16h bnxt = bcur;
      if (t < 7) bnxt = load_b_f16(k16, krow0 + (t + 1) * 16, H_, k0, lane);
      acc[t]     = __builtin_amdgcn_wmma_f32_16x16x32_f16(
          false, a0, false, bcur, (short)0, acc[t], false, false);
      acc[8 + t] = __builtin_amdgcn_wmma_f32_16x16x32_f16(
          false, a1, false, bcur, (short)0, acc[8 + t], false, false);
      bcur = bnxt;
    }
  }

  // scale by 1/sqrt(H)
#pragma unroll
  for (int t = 0; t < 16; ++t)
#pragma unroll
    for (int i = 0; i < 8; ++i) acc[t][i] *= INV_SQRT_H;

  // per-lane partial row max -> LDS (rows: i+8*hi and 16+i+8*hi)
#pragma unroll
  for (int i = 0; i < 8; ++i) {
    float pm0 = -3.402823466e38f, pm1 = -3.402823466e38f;
#pragma unroll
    for (int t = 0; t < 8; ++t) {
      pm0 = fmaxf(pm0, acc[t][i]);
      pm1 = fmaxf(pm1, acc[8 + t][i]);
    }
    red[(i + 8 * hi) * 256 + w * 32 + lane]      = pm0;
    red[(16 + i + 8 * hi) * 256 + w * 32 + lane] = pm1;
  }
  __syncthreads();
  if (tid < 32) {
    float m = -3.402823466e38f;
    for (int j = 0; j < 256; ++j) m = fmaxf(m, red[tid * 256 + j]);
    rmax[tid] = m;
  }
  __syncthreads();

  // exp + partial row sums
#pragma unroll
  for (int i = 0; i < 8; ++i) {
    const float mx0 = rmax[i + 8 * hi];
    const float mx1 = rmax[16 + i + 8 * hi];
    float s0 = 0.f, s1 = 0.f;
#pragma unroll
    for (int t = 0; t < 8; ++t) {
      const float e0 = __expf(acc[t][i] - mx0);
      const float e1 = __expf(acc[8 + t][i] - mx1);
      acc[t][i] = e0;     s0 += e0;
      acc[8 + t][i] = e1; s1 += e1;
    }
    red[(i + 8 * hi) * 256 + w * 32 + lane]      = s0;
    red[(16 + i + 8 * hi) * 256 + w * 32 + lane] = s1;
  }
  __syncthreads();
  if (tid < 32) {
    float s = 0.f;
    for (int j = 0; j < 256; ++j) s += red[tid * 256 + j];
    rsum[tid] = 1.0f / s;
  }
  __syncthreads();

  // normalize and write probs[b][m0+row][col]
#pragma unroll
  for (int i = 0; i < 8; ++i) {
    const int r0 = i + 8 * hi;
    const int r1 = 16 + i + 8 * hi;
    const float inv0 = rsum[r0];
    const float inv1 = rsum[r1];
    const long base0 = ((long)b * S_ + (m0 + r0)) * S_;
    const long base1 = ((long)b * S_ + (m0 + r1)) * S_;
#pragma unroll
    for (int t = 0; t < 8; ++t) {
      const int col = w * 128 + t * 16 + (lane & 15);
      probs[base0 + col] = acc[t][i] * inv0;
      probs[base1 + col] = acc[8 + t][i] * inv1;
    }
  }
}

// ---------------------------------------------------------------------------
// K3: out = probs @ v.  Block tile 64x256 (8 waves: 2(M) x 4(N), wave tile
// 32x64 = 8 accumulators). A = probs f32 -> f16 convert on load (2 frags,
// reused across 4 column tiles); B fragments software-pipelined and each
// feeds two wmma. K = 2048 in steps of 32.
// ---------------------------------------------------------------------------
__global__ __launch_bounds__(256)
void pv_gemm_kernel(const float* __restrict__ probs,
                    const _Float16* __restrict__ vT16,
                    float* __restrict__ out) {
  const int tid  = threadIdx.x;
  const int lane = tid & 31;
  const int w    = tid >> 5;
  const int hi   = (lane >> 4) & 1;
  const int b    = blockIdx.z;
  const int mb   = blockIdx.x * 64 + (w >> 2) * 32;   // row base within batch
  const int nb   = blockIdx.y * 256 + (w & 3) * 64;   // col base within H

  const long prow0 = (long)b * S_ + mb;               // probs rows ([B*S][S])
  const long vrow0 = (long)b * H_ + nb;               // vT rows ([B*H][S])

  v8f acc[8];                    // [t]: rows mb..+15 ; [4+t]: rows mb+16..+31
  const v8f vz = {};
#pragma unroll
  for (int t = 0; t < 8; ++t) acc[t] = vz;

  for (int k0 = 0; k0 < S_; k0 += 32) {
    const v16h a0 = load_a_f32cvt(probs, prow0,      S_, k0, lane);
    const v16h a1 = load_a_f32cvt(probs, prow0 + 16, S_, k0, lane);
    v16h bcur = load_b_f16(vT16, vrow0, S_, k0, lane);
#pragma unroll
    for (int t = 0; t < 4; ++t) {
      v16h bnxt = bcur;
      if (t < 3) bnxt = load_b_f16(vT16, vrow0 + (t + 1) * 16, S_, k0, lane);
      acc[t]     = __builtin_amdgcn_wmma_f32_16x16x32_f16(
          false, a0, false, bcur, (short)0, acc[t], false, false);
      acc[4 + t] = __builtin_amdgcn_wmma_f32_16x16x32_f16(
          false, a1, false, bcur, (short)0, acc[4 + t], false, false);
      bcur = bnxt;
    }
  }

#pragma unroll
  for (int i = 0; i < 8; ++i) {
    const long row0 = (long)b * S_ + mb + i + 8 * hi;
    const long row1 = row0 + 16;
#pragma unroll
    for (int t = 0; t < 4; ++t) {
      const int col = nb + t * 16 + (lane & 15);
      out[row0 * H_ + col] = acc[t][i];
      out[row1 * H_ + col] = acc[4 + t][i];
    }
  }
}

// ---------------------------------------------------------------------------
// Launcher
// ---------------------------------------------------------------------------
extern "C" void kernel_launch(void* const* d_in, const int* in_sizes, int n_in,
                              void* d_out, int out_size, void* d_ws, size_t ws_size,
                              hipStream_t stream) {
  const float* x  = (const float*)d_in[0];
  const float* Wq = (const float*)d_in[1];
  const float* bq = (const float*)d_in[2];
  const float* Wk = (const float*)d_in[3];
  const float* bk = (const float*)d_in[4];
  const float* Wv = (const float*)d_in[5];
  const float* bv = (const float*)d_in[6];
  const float* Wl = (const float*)d_in[7];

  float* out   = (float*)d_out;                       // [B,S,H]
  float* probs = out + (long)B_ * S_ * H_;            // [B,S,S]

  _Float16* q16  = (_Float16*)d_ws;                   // [B*S][H]
  _Float16* k16  = q16 + (long)B_ * S_ * H_;          // [B*S][H]
  _Float16* vT16 = k16 + (long)B_ * S_ * H_;          // [B][H][S]

  // K1: masked 25-tap projections (sparse mask -> stencil, not GEMM)
  proj_stencil_kernel<<<B_ * S_, 256, 0, stream>>>(
      x, Wq, bq, Wk, bk, Wv, bv, Wl, q16, k16, vT16);

  // K2: scores + softmax -> probs (32 rows x 1024 cols per block)
  scores_softmax_kernel<<<dim3(S_ / 32, B_), 256, 0, stream>>>(q16, k16, probs);

  // K3: out = probs @ v (64x256 block tile)
  pv_gemm_kernel<<<dim3(S_ / 64, H_ / 256, B_), 256, 0, stream>>>(probs, vT16, out);
}
